// ESN_18786186953267
// MI455X (gfx1250) — matrix operand order repaired and checked by reading, not compiled, not run on hardware
//
#include <hip/hip_runtime.h>
#include <math.h>

// Problem constants (from reference): B=512, T=512, I=64, R=1024
#define BB 512
#define TT 512
#define II 64
#define RR 1024

// Step-kernel tiling: 128 (M=batch) x 64 (N=reservoir) per workgroup,
// 256 threads = 8 wave32 arranged 4 (M) x 2 (N); each wave owns 32x32.
#define BM 128
#define BN 64
#define BK 32
#define NKW (RR / BK)             // 32 chunks of W/h
#define KCHUNKS ((RR + II) / BK)  // 34 total (2 extra chunks = input projection)

typedef __attribute__((ext_vector_type(16))) __bf16 bf16x16;
typedef __attribute__((ext_vector_type(8)))  float  f32x8;
typedef unsigned int u32x4 __attribute__((ext_vector_type(4)));
typedef int          i32x4 __attribute__((ext_vector_type(4)));
typedef int          i32x8 __attribute__((ext_vector_type(8)));
typedef int          v4i   __attribute__((vector_size(16)));  // builtin arg type

union FragU { uint4 q[2]; bf16x16 v; };

// ---- CDNA5 feature probes (compile-safe fallbacks) -------------------------
#if defined(__AMDGCN__) && __has_builtin(__builtin_amdgcn_global_load_async_to_lds_b128)
#define ASYNC_G2L 1
#else
#define ASYNC_G2L 0
#endif

#if defined(__AMDGCN__) && __has_builtin(__builtin_amdgcn_tensor_load_to_lds) && \
    __has_builtin(__builtin_amdgcn_s_wait_tensorcnt)
#define HAVE_TDM 1
#else
#define HAVE_TDM 0
#endif

#if ASYNC_G2L && __has_builtin(__builtin_amdgcn_s_wait_asynccnt)
#define WAIT_ASYNC(n) __builtin_amdgcn_s_wait_asynccnt(n)
#else
#define WAIT_ASYNC(n) ((void)0)
#endif

#if HAVE_TDM
#define WAIT_TENSOR(n) __builtin_amdgcn_s_wait_tensorcnt(n)
#define ASY_PER_CHUNK 2   // per wave: 2 async A-tile loads per chunk (B via TDM)
#else
#define WAIT_TENSOR(n) ((void)0)
#define ASY_PER_CHUNK 3   // per wave: 2 A + 1 B async loads per chunk
#endif

__device__ __forceinline__ unsigned int f2bf_raw(float f) {
  union { float f; unsigned int u; } x; x.f = f;
  return (x.u + 0x7FFFu + ((x.u >> 16) & 1u)) >> 16;  // RNE f32->bf16
}
__device__ __forceinline__ float bf2f(unsigned short v) {
  union { unsigned int u; float f; } x; x.u = ((unsigned int)v) << 16;
  return x.f;
}
__device__ __forceinline__ float fast_tanh(float x) {
#if __has_builtin(__builtin_amdgcn_tanhf)
  return __builtin_amdgcn_tanhf(x);
#elif __has_builtin(__builtin_amdgcn_tanh_f32)
  return __builtin_amdgcn_tanh_f32(x);
#else
  return tanhf(x);
#endif
}

// Per-lane global->LDS 16B copy: async on CDNA5, sync fallback otherwise.
__device__ __forceinline__ void g2l_b128(uint4* lds, const void* g) {
#if ASYNC_G2L
  __builtin_amdgcn_global_load_async_to_lds_b128(
      (__attribute__((address_space(1))) v4i*)g,
      (__attribute__((address_space(3))) v4i*)lds, 0, 0);
#else
  *lds = *(const uint4*)g;
#endif
}

#if HAVE_TDM
__device__ __forceinline__ unsigned lds_addr_of(const void* p) {
  return (unsigned)(size_t)(__attribute__((address_space(3))) const void*)p;
}
// 2D TDM tile load: rows x cols bf16 tile from a (tensor_w-wide) bf16 tensor,
// packed row-major into LDS. D# layout per CDNA5 ISA 8.3/8.4.
__device__ __forceinline__ void tdm_load_tile(const unsigned short* gtile,
                                              unsigned lds_byte_addr,
                                              unsigned tensor_w, int rows,
                                              int cols) {
  unsigned long long ga = (unsigned long long)(size_t)gtile;
  u32x4 g0 = (u32x4)0;
  g0[0] = 1u;                                   // count=1 (valid user D#)
  g0[1] = lds_byte_addr;                        // lds_addr  [63:32]
  g0[2] = (unsigned)(ga & 0xFFFFFFFFu);         // global_addr[95:64]
  g0[3] = (unsigned)((ga >> 32) & 0x01FFFFFFu)  // global_addr[120:96]
          | (2u << 30);                         // type=2 ("image") [127:126]
  i32x8 g1 = (i32x8)0;
  g1[0] = 1 << 16;                                        // data_size=1 (2B)
  g1[1] = (int)((tensor_w & 0xFFFFu) << 16);              // tensor_dim0 lo16
  g1[2] = (int)(((tensor_w >> 16) & 0xFFFFu) |            // tensor_dim0 hi16
                ((unsigned)RR << 16));                    // tensor_dim1 lo16
  g1[3] = (int)(((unsigned)cols & 0xFFFFu) << 16);        // tile_dim0
  g1[4] = rows & 0xFFFF;                                  // tile_dim1
  g1[5] = (int)tensor_w;                                  // tensor_dim0_stride
  i32x4 z = (i32x4)0;
#if __clang_major__ >= 23
  __builtin_amdgcn_tensor_load_to_lds(g0, g1, z, z, (i32x8)0, 0);
#else
  __builtin_amdgcn_tensor_load_to_lds(g0, g1, z, z, 0);
#endif
}
#endif  // HAVE_TDM

// ---------------------------------------------------------------------------
__global__ __launch_bounds__(256) void esn_f32_to_bf16(
    const float* __restrict__ src, unsigned short* __restrict__ dst, int n) {
  int i = blockIdx.x * 256 + threadIdx.x;
  if (i < n) dst[i] = (unsigned short)f2bf_raw(src[i]);
}

__global__ __launch_bounds__(256) void esn_zero_u32(
    unsigned int* __restrict__ p, int n_u32) {
  int i = blockIdx.x * 256 + threadIdx.x;
  if (i < n_u32) p[i] = 0u;
}

// ---------------------------------------------------------------------------
// Stage one K-chunk (A: 128x32 h/x tile, B: 64x32 W/Win tile) into LDS buffer.
__device__ __forceinline__ void load_tiles(
    int kb, int buf, int tid, int waveid, int m0, int n0, int t,
    const unsigned short* __restrict__ h_in,
    const unsigned short* __restrict__ Xb,
    const unsigned short* __restrict__ Wb,
    const unsigned short* __restrict__ Winb,
    uint4 (*Ash4)[(BM * BK) / 8], uint4 (*Bsh4)[(BN * BK) / 8]) {
  // ---- A tile: 512 16B segments, 2 per thread ----
  int s0 = tid, s1 = tid + 256;
  int r0 = s0 >> 2, k0 = (s0 & 3) * 8;
  int r1 = s1 >> 2, k1 = (s1 & 3) * 8;
  const unsigned short *pa0, *pa1;
  if (kb < NKW) {
    pa0 = h_in + (size_t)(m0 + r0) * RR + kb * BK + k0;
    pa1 = h_in + (size_t)(m0 + r1) * RR + kb * BK + k1;
  } else {
    int kk = (kb - NKW) * BK;
    pa0 = Xb + ((size_t)(m0 + r0) * TT + t) * II + kk + k0;
    pa1 = Xb + ((size_t)(m0 + r1) * TT + t) * II + kk + k1;
  }
  g2l_b128(&Ash4[buf][s0], pa0);
  g2l_b128(&Ash4[buf][s1], pa1);

  // ---- B tile ----
#if HAVE_TDM
  if (waveid == 0) {
    if (kb < NKW)
      tdm_load_tile(Wb + (size_t)n0 * RR + kb * BK, lds_addr_of(&Bsh4[buf][0]),
                    RR, BN, BK);
    else
      tdm_load_tile(Winb + (size_t)n0 * II + (kb - NKW) * BK,
                    lds_addr_of(&Bsh4[buf][0]), II, BN, BK);
  }
#else
  int rb = tid >> 2, kB = (tid & 3) * 8;
  const unsigned short* pb =
      (kb < NKW) ? Wb + (size_t)(n0 + rb) * RR + kb * BK + kB
                 : Winb + (size_t)(n0 + rb) * II + (kb - NKW) * BK + kB;
  g2l_b128(&Bsh4[buf][tid], pb);
#endif
}

// ---------------------------------------------------------------------------
// One ESN timestep: h_out = tanh( x_t @ Win^T + h_in @ W^T ), bf16 WMMA,
// double-buffered LDS with async / TDM staging.
// ---------------------------------------------------------------------------
__global__ __launch_bounds__(256) void esn_step(
    const unsigned short* __restrict__ Xb,    // [B,T,I] bf16
    const unsigned short* __restrict__ Wb,    // [R,R]  bf16
    const unsigned short* __restrict__ Winb,  // [R,I]  bf16
    const unsigned short* __restrict__ h_in,  // [B,R]  bf16
    unsigned short*       __restrict__ h_out, // [B,R]  bf16
    int t) {
  __shared__ uint4 Ash4[2][(BM * BK) / 8];  // 2 x 8 KB
  __shared__ uint4 Bsh4[2][(BN * BK) / 8];  // 2 x 4 KB

  const int tid    = threadIdx.x;
  const int lane   = tid & 31;
  const int l15    = lane & 15;
  const int lhalf  = lane >> 4;
  const int waveid = tid >> 5;
  const int wave_m = waveid & 3;   // 4 waves x 32 rows
  const int wave_n = waveid >> 2;  // 2 waves x 32 cols

  const int m0 = blockIdx.y * BM;
  const int n0 = blockIdx.x * BN;

  f32x8 acc[2][2];
#pragma unroll
  for (int mm = 0; mm < 2; ++mm)
#pragma unroll
    for (int nn = 0; nn < 2; ++nn) acc[mm][nn] = (f32x8)0.0f;

  load_tiles(0, 0, tid, waveid, m0, n0, t, h_in, Xb, Wb, Winb, Ash4, Bsh4);

  for (int kb = 0; kb < KCHUNKS; ++kb) {
    const int cur = kb & 1;
    if (kb + 1 < KCHUNKS) {
      // prefetch next chunk into the other buffer, then wait for current
      load_tiles(kb + 1, cur ^ 1, tid, waveid, m0, n0, t, h_in, Xb, Wb, Winb,
                 Ash4, Bsh4);
      WAIT_ASYNC(ASY_PER_CHUNK);
      if (waveid == 0) WAIT_TENSOR(1);
    } else {
      WAIT_ASYNC(0);
      if (waveid == 0) WAIT_TENSOR(0);
    }
    __syncthreads();  // current buffer complete for all waves

    const unsigned int* Ashu = reinterpret_cast<const unsigned int*>(Ash4[cur]);
    const unsigned int* Bshu = reinterpret_cast<const unsigned int*>(Bsh4[cur]);

    // per-wave fragments matching the ISA 16-bit A / B WMMA VGPR layouts
    FragU fa[2], fb[2];
#pragma unroll
    for (int mm = 0; mm < 2; ++mm) {
      int row = wave_m * 32 + mm * 16 + l15;
      int dw = row * (BK / 2) + lhalf * 4;
      fa[mm].q[0] = *reinterpret_cast<const uint4*>(Ashu + dw);
      fa[mm].q[1] = *reinterpret_cast<const uint4*>(Ashu + dw + 8);
    }
#pragma unroll
    for (int nn = 0; nn < 2; ++nn) {
      int nrow = wave_n * 32 + nn * 16 + l15;
      int dw = nrow * (BK / 2) + lhalf * 4;
      fb[nn].q[0] = *reinterpret_cast<const uint4*>(Bshu + dw);
      fb[nn].q[1] = *reinterpret_cast<const uint4*>(Bshu + dw + 8);
    }

#pragma unroll
    for (int mm = 0; mm < 2; ++mm)
#pragma unroll
      for (int nn = 0; nn < 2; ++nn)
        acc[mm][nn] = __builtin_amdgcn_wmma_f32_16x16x32_bf16(
            false, fa[mm].v, false, fb[nn].v, (short)0, acc[mm][nn], false,
            false);

    __syncthreads();  // all waves done reading before this buffer is refilled
  }

  // ---- epilogue: tanh + bf16 store of new state ----
#pragma unroll
  for (int mm = 0; mm < 2; ++mm) {
#pragma unroll
    for (int nn = 0; nn < 2; ++nn) {
      int n = n0 + wave_n * 32 + nn * 16 + l15;
#pragma unroll
      for (int j = 0; j < 8; ++j) {
        int m = m0 + wave_m * 32 + mm * 16 + lhalf * 8 + j;
        float v = fast_tanh(acc[mm][nn][j]);
        h_out[(size_t)m * RR + n] = (unsigned short)f2bf_raw(v);
      }
    }
  }
}

// ---------------------------------------------------------------------------
__global__ __launch_bounds__(256) void esn_readout(
    const unsigned short* __restrict__ h, const float* __restrict__ Wout_w,
    const float* __restrict__ Wout_b, float* __restrict__ out) {
  __shared__ float red[256];
  int b = blockIdx.x;
  float s = 0.0f;
  for (int r = threadIdx.x; r < RR; r += 256)
    s += bf2f(h[(size_t)b * RR + r]) * Wout_w[r];
  red[threadIdx.x] = s;
  __syncthreads();
  for (int off = 128; off > 0; off >>= 1) {
    if (threadIdx.x < off) red[threadIdx.x] += red[threadIdx.x + off];
    __syncthreads();
  }
  if (threadIdx.x == 0) out[b] = red[0] + Wout_b[0];
}

// ---------------------------------------------------------------------------
extern "C" void kernel_launch(void* const* d_in, const int* in_sizes, int n_in,
                              void* d_out, int out_size, void* d_ws,
                              size_t ws_size, hipStream_t stream) {
  const float* X      = (const float*)d_in[0];  // [512,512,64]
  const float* Win    = (const float*)d_in[1];  // [1024,64]
  const float* W      = (const float*)d_in[2];  // [1024,1024]
  const float* Wout_w = (const float*)d_in[3];  // [1,1024]
  const float* Wout_b = (const float*)d_in[4];  // [1]
  float* out = (float*)d_out;                   // [512]

  // workspace layout (bf16 buffers): W 2MiB | Win 128KiB | h0 1MiB | h1 1MiB | X 32MiB
  unsigned short* Wb   = (unsigned short*)d_ws;
  unsigned short* Winb = Wb + (size_t)RR * RR;
  unsigned short* h0   = Winb + (size_t)RR * II;
  unsigned short* h1   = h0 + (size_t)BB * RR;
  unsigned short* Xb   = h1 + (size_t)BB * RR;

  esn_f32_to_bf16<<<(RR * RR + 255) / 256, 256, 0, stream>>>(W, Wb, RR * RR);
  esn_f32_to_bf16<<<(RR * II + 255) / 256, 256, 0, stream>>>(Win, Winb,
                                                             RR * II);
  esn_f32_to_bf16<<<(BB * TT * II + 255) / 256, 256, 0, stream>>>(
      X, Xb, BB * TT * II);
  esn_zero_u32<<<(BB * RR / 2 + 255) / 256, 256, 0, stream>>>(
      (unsigned int*)h0, BB * RR / 2);

  // strictly sequential scan: one launch per timestep
  dim3 grid(RR / BN, BB / BM);  // (16, 4) = 64 workgroups
  unsigned short* hin = h0;
  unsigned short* hout = h1;
  for (int t = 0; t < TT; ++t) {
    esn_step<<<grid, 256, 0, stream>>>(Xb, Wb, Winb, hin, hout, t);
    unsigned short* tmp = hin; hin = hout; hout = tmp;
  }

  esn_readout<<<BB, 256, 0, stream>>>(hin, Wout_w, Wout_b, out);
}
